// LSTMNet_13786845020606
// MI455X (gfx1250) — matrix operand order repaired and checked by reading, not compiled, not run on hardware
//
#include <hip/hip_runtime.h>

// ---------------------------------------------------------------------------
// Persistent 2-layer LSTM for MI455X (gfx1250, wave32, WMMA bf16).
// 32 WGs x 16 batch rows, no grid sync.  1024 threads (32 waves) per WG;
// each wave owns ONE 16-wide hidden tile across the 4 gates.
// An opaque per-iteration zero offset keeps LICM from hoisting weight loads
// out of the time loop while preserving global (AS1) addressing.
// ---------------------------------------------------------------------------

#define LAYERS 2
#define H      512
#define IN     32
#define OUT    8
#define BATCH  512
#define SEQ    128
#define FUT    48
#define TF     24

#define K0   (IN + H)        // 544  -> 17 k-chunks of 32
#define K1   (2 * H)         // 1024 -> 32 k-chunks of 32
#define KC0  (K0 / 32)       // 17
#define KC1  (K1 / 32)       // 32
#define NT   (4 * H / 16)    // 128 n-tiles of 16 gate columns
#define K0P  552             // padded LDS row (multiple of 8 -> 16B aligned rows)
#define K1P  1032
#define THREADS 1024

typedef __attribute__((ext_vector_type(8)))  float  v8f;
typedef __attribute__((ext_vector_type(8)))  __bf16 v8bf;
typedef __attribute__((ext_vector_type(16))) __bf16 v16bf;

#define N0_ELEMS (NT * KC0 * 512)   // 1,114,112 bf16 (layer0 swizzled weights)
#define N1_ELEMS (NT * KC1 * 512)   // 2,097,152 bf16 (layer1 swizzled weights)
#define WSW0_BYTES ((size_t)N0_ELEMS * 2)
#define WSW1_BYTES ((size_t)N1_ELEMS * 2)

__device__ __forceinline__ float fast_sigmoid(float x) {
    return 1.0f / (1.0f + __expf(-x));
}
__device__ __forceinline__ float fast_tanh(float x) {
    float e = __expf(2.0f * x);
    return (e - 1.0f) / (e + 1.0f);
}

// ---------------------------------------------------------------------------
// Prep: concat [W_ih | W_hh] along K, convert to bf16, and swizzle into the
// per-lane B order this code uses for v_wmma_f32_16x16x32_bf16 (contiguous k):
//   fragment element (lane, slot):  k = kc*32 + (lane>>4)*16 + slot,
//                                   n = ntile*16 + (lane&15)
// stored contiguously: ((ntile*KC + kc)*32 + lane)*16 + slot.
// Also precompute bias2[l][n] = b_ih[n] + b_hh[n].
// ---------------------------------------------------------------------------
__global__ void pack_params(const float* __restrict__ w_ih0, const float* __restrict__ w_hh0,
                            const float* __restrict__ b_ih0, const float* __restrict__ b_hh0,
                            const float* __restrict__ w_ih1, const float* __restrict__ w_hh1,
                            const float* __restrict__ b_ih1, const float* __restrict__ b_hh1,
                            __bf16* __restrict__ wsw0, __bf16* __restrict__ wsw1,
                            float* __restrict__ bias2)
{
    int idx = blockIdx.x * blockDim.x + threadIdx.x;
    if (idx < N0_ELEMS) {
        int slot = idx & 15, lane = (idx >> 4) & 31;
        int ck = idx >> 9;
        int kc = ck % KC0, ntile = ck / KC0;
        int k = kc * 32 + ((lane >> 4) << 4) + slot;
        int n = ntile * 16 + (lane & 15);
        float v = (k < IN) ? w_ih0[n * IN + k] : w_hh0[n * H + (k - IN)];
        wsw0[idx] = (__bf16)v;
    } else if (idx < N0_ELEMS + N1_ELEMS) {
        int j = idx - N0_ELEMS;
        int slot = j & 15, lane = (j >> 4) & 31;
        int ck = j >> 9;
        int kc = ck % KC1, ntile = ck / KC1;
        int k = kc * 32 + ((lane >> 4) << 4) + slot;
        int n = ntile * 16 + (lane & 15);
        float v = (k < H) ? w_ih1[n * H + k] : w_hh1[n * H + (k - H)];
        wsw1[j] = (__bf16)v;
    } else if (idx < N0_ELEMS + N1_ELEMS + 2 * 4 * H) {
        int j = idx - N0_ELEMS - N1_ELEMS;
        int l = j >> 11, n = j & (4 * H - 1);
        bias2[j] = l ? (b_ih1[n] + b_hh1[n]) : (b_ih0[n] + b_hh0[n]);
    }
}

// ---------------------------------------------------------------------------
// Gate GEMM for one layer: [16 x K] (LDS bf16) @ [K x 2048] (swizzled bf16).
// Wave w owns hidden tile w (16 cols) for all 4 gates -> 4 accumulators.
// A fragment contiguous (matches B packing): lanes 0-15 K={0..15}, lanes 16-31
// K={16..31} -> two ds_load_b128 per k-chunk.  B double-buffered.
// ---------------------------------------------------------------------------
template <int KC, int KP>
__device__ __forceinline__ void gate_matmul(const __bf16* xb,
                                            const __bf16* wsw,
                                            int wave, int lane, v8f acc[4])
{
    const int m  = lane & 15;
    const int kb = (lane >> 4) << 3;   // 0 or 8
    const __bf16* arow  = xb + m * KP;
    const __bf16* wlane = wsw + (lane << 4);

    v16bf Bcur[4];
    #pragma unroll
    for (int g = 0; g < 4; ++g)
        Bcur[g] = *(const v16bf*)(wlane + ((size_t)((g * 32 + wave) * KC) << 9));

    for (int kc = 0; kc < KC; ++kc) {
        v16bf Bnext[4];
        if (kc + 1 < KC) {
            #pragma unroll
            for (int g = 0; g < 4; ++g)
                Bnext[g] = *(const v16bf*)(wlane +
                    ((size_t)((g * 32 + wave) * KC + kc + 1) << 9));
        }
        v8bf alo = *(const v8bf*)(arow + kc * 32 + 2 * kb);
        v8bf ahi = *(const v8bf*)(arow + kc * 32 + 2 * kb + 8);
        v16bf A = __builtin_shufflevector(alo, ahi,
                   0, 1, 2, 3, 4, 5, 6, 7, 8, 9, 10, 11, 12, 13, 14, 15);
        #pragma unroll
        for (int g = 0; g < 4; ++g)
            acc[g] = __builtin_amdgcn_wmma_f32_16x16x32_bf16(
                false, A, false, Bcur[g], (short)0, acc[g], false, false);
        if (kc + 1 < KC) {
            #pragma unroll
            for (int g = 0; g < 4; ++g) Bcur[g] = Bnext[g];
        }
    }
}

// LSTM pointwise (gate order i,f,g,o) on C/D fragment layout:
// VGPR r, lane l -> row M = (l>=16)*8 + r, col N = (l&15).
__device__ __forceinline__ void lstm_pointwise_store(
    v8f acc[4], v8f& cst, const float biasr[4],
    __bf16* dst0, int stride0, __bf16* dst1, int stride1,
    int wave, int lane)
{
    const int n_l  = lane & 15;
    const int mrow = (lane >> 4) << 3;
    const int hcol = wave * 16 + n_l;
    #pragma unroll
    for (int r = 0; r < 8; ++r) {
        float iv = fast_sigmoid(acc[0][r] + biasr[0]);
        float fv = fast_sigmoid(acc[1][r] + biasr[1]);
        float gv = fast_tanh   (acc[2][r] + biasr[2]);
        float ov = fast_sigmoid(acc[3][r] + biasr[3]);
        float c2 = fv * cst[r] + iv * gv;
        cst[r] = c2;
        __bf16 hb = (__bf16)(ov * fast_tanh(c2));
        dst0[(mrow + r) * stride0 + hcol] = hb;
        if (dst1) dst1[(mrow + r) * stride1 + hcol] = hb;
    }
}

// ---------------------------------------------------------------------------
// Persistent kernel: 32 blocks x 1024 threads (32 waves); each block owns 16
// batch rows for all 175 timesteps.  Only workgroup barriers.
// ---------------------------------------------------------------------------
__global__ __launch_bounds__(THREADS, 1)
void lstm_persistent(const float* __restrict__ x,
                     const float* __restrict__ future_time,
                     const float* __restrict__ h0,
                     const float* __restrict__ c0,
                     const float* __restrict__ fc_w_in,
                     const float* __restrict__ fc_b,
                     const __bf16* __restrict__ wsw0_in,
                     const __bf16* __restrict__ wsw1_in,
                     const float* __restrict__ bias2,
                     float* __restrict__ out)
{
    __shared__ __align__(16) __bf16 xbuf0[16 * K0P]; // [x_t | h_layer0_prev]
    __shared__ __align__(16) __bf16 xbuf1[16 * K1P]; // [h_layer0_new | h_layer1_prev]
    __shared__ float ybuf[16 * OUT];

    const int tid   = threadIdx.x;
    const int lane  = tid & 31;
    const int wave  = tid >> 5;      // 0..31 == hidden tile index
    const int bbase = blockIdx.x * 16;
    const int n_l   = lane & 15;

    // broadcast initial hidden state into LDS (bf16)
    for (int i = tid; i < 16 * H; i += THREADS) {
        int m = i >> 9, hc = i & (H - 1);
        xbuf0[m * K0P + IN + hc] = (__bf16)h0[hc];
        xbuf1[m * K1P + H  + hc] = (__bf16)h0[H + hc];
    }

    // persistent cell state in registers + preloaded gate biases
    v8f   cst[2];
    float biasr[2][4];
    #pragma unroll
    for (int l = 0; l < 2; ++l) {
        int hcol = wave * 16 + n_l;
        float cv = c0[l * H + hcol];
        #pragma unroll
        for (int r = 0; r < 8; ++r) cst[l][r] = cv;
        #pragma unroll
        for (int g = 0; g < 4; ++g)
            biasr[l][g] = bias2[l * 4 * H + (g * 32 + wave) * 16 + n_l];
    }
    __syncthreads();

    const int TOTAL = SEQ + FUT - 1; // 175
    for (int step = 0; step < TOTAL; ++step) {
        // Opaque zero offset, redefined every iteration: the weight-load
        // addresses contain a loop-defined value so LICM cannot hoist them
        // out of the time loop, yet the pointer base keeps its global-AS
        // provenance (global_load_b128, LOADcnt only — no FLAT/DScnt).
        size_t zoff = 0;
        asm volatile("" : "+s"(zoff));
        const __bf16* wsw0 = wsw0_in + zoff;
        const __bf16* wsw1 = wsw1_in + zoff;
        const float*  fc_w = fc_w_in + zoff;

        // ---- build x_t into xbuf0[:, 0:IN) : 512 of 1024 threads ----
        if (tid < 16 * IN) {
            int m = tid >> 5, c = tid & 31;
            float v;
            if (step < SEQ) {
                v = x[((size_t)step * BATCH + bbase + m) * IN + c];
            } else {
                v = (c < OUT) ? ybuf[m * OUT + c]
                              : future_time[((size_t)(step - SEQ) * BATCH + bbase + m) * TF + (c - OUT)];
            }
            xbuf0[m * K0P + c] = (__bf16)v;
        }
        __syncthreads();

        // ---- layer 0: gates = [x_t | h0_prev] @ Wcat0^T ----
        {
            v8f z = {0.f, 0.f, 0.f, 0.f, 0.f, 0.f, 0.f, 0.f};
            v8f acc[4] = {z, z, z, z};
            gate_matmul<KC0, K0P>(xbuf0, wsw0, wave, lane, acc);
            __syncthreads(); // all reads of xbuf0 done before h rewrite
            lstm_pointwise_store(acc, cst[0], biasr[0],
                                 xbuf1, K1P,            // -> layer1 input cols [0,H)
                                 xbuf0 + IN, K0P,       // -> next-step h_prev
                                 wave, lane);
        }
        __syncthreads();

        // ---- layer 1: gates = [h0_new | h1_prev] @ Wcat1^T ----
        {
            v8f z = {0.f, 0.f, 0.f, 0.f, 0.f, 0.f, 0.f, 0.f};
            v8f acc[4] = {z, z, z, z};
            gate_matmul<KC1, K1P>(xbuf1, wsw1, wave, lane, acc);
            __syncthreads(); // reads of h1_prev done before rewrite
            lstm_pointwise_store(acc, cst[1], biasr[1],
                                 xbuf1 + H, K1P,        // -> h1 (FC input / next prev)
                                 (__bf16*)nullptr, 0,
                                 wave, lane);
        }
        __syncthreads();

        // ---- FC head: y = h1 @ fc_w^T + fc_b  (last encoder + every decode) ----
        if (step >= SEQ - 1) {
            if (tid < 16 * OUT) {
                int m = tid >> 3, o = tid & 7;
                float s = fc_b[o];
                const __bf16* hrow = xbuf1 + m * K1P + H;
                const float*  wr   = fc_w + o * H;
                #pragma unroll 8
                for (int k = 0; k < H; ++k) s += (float)hrow[k] * wr[k];
                int ystep = step - (SEQ - 1);
                ybuf[m * OUT + o] = s;
                out[((size_t)ystep * BATCH + bbase + m) * OUT + o] = s;
            }
            __syncthreads();
        }
    }
}

extern "C" void kernel_launch(void* const* d_in, const int* in_sizes, int n_in,
                              void* d_out, int out_size, void* d_ws, size_t ws_size,
                              hipStream_t stream) {
    const float* x           = (const float*)d_in[0];
    const float* future_time = (const float*)d_in[1];
    const float* h0          = (const float*)d_in[2];
    const float* c0          = (const float*)d_in[3];
    const float* w_ih0       = (const float*)d_in[4];
    const float* w_hh0       = (const float*)d_in[5];
    const float* b_ih0       = (const float*)d_in[6];
    const float* b_hh0       = (const float*)d_in[7];
    const float* w_ih1       = (const float*)d_in[8];
    const float* w_hh1       = (const float*)d_in[9];
    const float* b_ih1       = (const float*)d_in[10];
    const float* b_hh1       = (const float*)d_in[11];
    const float* fc_w        = (const float*)d_in[12];
    const float* fc_b        = (const float*)d_in[13];

    char*   ws    = (char*)d_ws;
    __bf16* wsw0  = (__bf16*)ws;
    __bf16* wsw1  = (__bf16*)(ws + WSW0_BYTES);
    float*  bias2 = (float*)(ws + WSW0_BYTES + WSW1_BYTES);

    int total = N0_ELEMS + N1_ELEMS + 2 * 4 * H;
    pack_params<<<(total + 255) / 256, 256, 0, stream>>>(
        w_ih0, w_hh0, b_ih0, b_hh0, w_ih1, w_hh1, b_ih1, b_hh1,
        wsw0, wsw1, bias2);

    lstm_persistent<<<BATCH / 16, THREADS, 0, stream>>>(
        x, future_time, h0, c0, fc_w, fc_b, wsw0, wsw1, bias2,
        (float*)d_out);
}
